// MultiHeadAttention_59450937311237
// MI455X (gfx1250) — matrix operand order repaired and checked by reading, not compiled, not run on hardware
//
#include <hip/hip_runtime.h>
#include <hip/hip_bf16.h>

#define BS   4
#define SEQ  2048
#define DM   1024
#define NH   16
#define DKH  64
#define ROWS (BS*SEQ)      // 8192
#define NEGV (-1e30f)

typedef unsigned int  u32;
typedef unsigned short u16;
typedef __attribute__((ext_vector_type(4)))  u32    u32x4;
typedef __attribute__((ext_vector_type(16))) __bf16 v16bf;
typedef __attribute__((ext_vector_type(8)))  float  v8f;

union FragBF {
    v16bf bf;
    u32x4 q[2];
};

__device__ __forceinline__ u16 f2bf(float f) {
    u32 u = __float_as_uint(f);
    u32 r = (u + 0x7FFFu + ((u >> 16) & 1u)) >> 16;  // round-to-nearest-even
    return (u16)r;
}

__device__ __forceinline__ v8f wmma_bf16(const FragBF& A, const FragBF& B, v8f C) {
    return __builtin_amdgcn_wmma_f32_16x16x32_bf16(
        false, A.bf, false, B.bf, (short)0, C, false, false);
}

// A fragment (16x32 bf16): lanes 0-15 hold K=0..15, lanes 16-31 K=16..31
__device__ __forceinline__ void load_a(FragBF& a, const u16* __restrict__ arow, int k, int half) {
    a.q[0] = *(const u32x4*)(arow + k + half * 16);
    a.q[1] = *(const u32x4*)(arow + k + half * 16 + 8);
}
// B fragment (32x16 bf16): lane = N%16, contiguous 16 K starting at 16*(lane/16); brow already has +half*16
__device__ __forceinline__ void load_b(FragBF& b, const u16* __restrict__ brow, int k) {
    b.q[0] = *(const u32x4*)(brow + k);
    b.q[1] = *(const u32x4*)(brow + k + 8);
}

// ---------------- prep kernels ----------------

__global__ void cvt_bf16_kernel(const float* __restrict__ src, u16* __restrict__ dst, int n) {
    int i = blockIdx.x * blockDim.x + threadIdx.x;
    if (i < n) dst[i] = f2bf(src[i]);
}

// src: fp32 [K=1024][N=1024]  ->  dst: bf16 [N=1024][K=1024]
__global__ void transpose_w_kernel(const float* __restrict__ src, u16* __restrict__ dst) {
    int i = blockIdx.x * blockDim.x + threadIdx.x;   // over 1M elements
    int n = i >> 10;
    int k = i & 1023;
    dst[i] = f2bf(src[k * DM + n]);
}

// ---------------- GEMM: Y = X(bf16 [ROWS][1024]) * Wt(bf16 [N][K]) + bias ----------------
// wave tile 32x64 (2 M-frags x 4 N-frags, 8 WMMA per K-step), register double-buffered.
// block = 8 waves -> 128x128 block tile. grid = (ROWS/128, DM/128).
// mode 0/1: bf16 out [b][h][l][dk] (Q,K); mode 2: bf16 out [b][h][dk][l] (Vt); mode 3: fp32 row-major.
__global__ void gemm_kernel(const u16* __restrict__ X, const u16* __restrict__ Wt,
                            const float* __restrict__ bias, void* __restrict__ out, int mode) {
    const int lane = threadIdx.x & 31;
    const int wave = threadIdx.x >> 5;
    const int half = lane >> 4;
    const int l16  = lane & 15;
    const int Mbase = blockIdx.x * 128 + (wave & 3) * 32;
    const int Nbase = blockIdx.y * 128 + (wave >> 2) * 64;

    v8f acc[2][4];
#pragma unroll
    for (int m = 0; m < 2; ++m)
#pragma unroll
        for (int f = 0; f < 4; ++f) acc[m][f] = {};

    const u16* arow[2];
    arow[0] = X + (size_t)(Mbase + l16) * DM;
    arow[1] = X + (size_t)(Mbase + 16 + l16) * DM;
    const u16* brow[4];
#pragma unroll
    for (int f = 0; f < 4; ++f)
        brow[f] = Wt + (size_t)(Nbase + f * 16 + l16) * DM + half * 16;

    FragBF A0[2], B0[4], A1[2], B1[4];
#pragma unroll
    for (int m = 0; m < 2; ++m) load_a(A0[m], arow[m], 0, half);
#pragma unroll
    for (int f = 0; f < 4; ++f) load_b(B0[f], brow[f], 0);

    for (int k = 0; k < DM; k += 64) {
        // prefetch k+32
#pragma unroll
        for (int m = 0; m < 2; ++m) load_a(A1[m], arow[m], k + 32, half);
#pragma unroll
        for (int f = 0; f < 4; ++f) load_b(B1[f], brow[f], k + 32);
#pragma unroll
        for (int f = 0; f < 4; ++f)
#pragma unroll
            for (int m = 0; m < 2; ++m)
                acc[m][f] = wmma_bf16(A0[m], B0[f], acc[m][f]);
        // prefetch k+64
        if (k + 64 < DM) {
#pragma unroll
            for (int m = 0; m < 2; ++m) load_a(A0[m], arow[m], k + 64, half);
#pragma unroll
            for (int f = 0; f < 4; ++f) load_b(B0[f], brow[f], k + 64);
        }
#pragma unroll
        for (int f = 0; f < 4; ++f)
#pragma unroll
            for (int m = 0; m < 2; ++m)
                acc[m][f] = wmma_bf16(A1[m], B1[f], acc[m][f]);
    }

#pragma unroll
    for (int m = 0; m < 2; ++m)
#pragma unroll
        for (int f = 0; f < 4; ++f) {
            int n = Nbase + f * 16 + l16;
            float bvv = bias[n];
#pragma unroll
            for (int v = 0; v < 8; ++v) {
                int r = Mbase + m * 16 + half * 8 + v;
                float val = acc[m][f][v] + bvv;
                if (mode == 3) {
                    ((float*)out)[(size_t)r * DM + n] = val;
                } else {
                    int b  = r >> 11, l = r & 2047;
                    int h  = n >> 6,  d = n & 63;
                    u16* o = (u16*)out;
                    if (mode == 2)
                        o[(((size_t)(b * NH + h)) * DKH + d) * SEQ + l] = f2bf(val);
                    else
                        o[(((size_t)(b * NH + h)) * SEQ + l) * DKH + d] = f2bf(val);
                }
            }
        }
}

// ---------------- flash attention ----------------
// block = 8 waves, each wave owns 16 queries of one (b,h); K/V tiles staged in LDS per block.
__global__ void flash_kernel(const u16* __restrict__ Qb, const u16* __restrict__ Kb,
                             const u16* __restrict__ Vtb, const int* __restrict__ mask,
                             u16* __restrict__ AO) {
    __shared__ __align__(16) u16 Ks[32][72];        // 32 keys x 64 feats, padded row (conflict-free)
    __shared__ __align__(16) u16 Vs[64][40];        // 64 dk x 32 keys, padded row
    __shared__ __align__(16) u16 pst[8][16][40];    // per-wave 16x32 P staging, padded row
    const int tid  = threadIdx.x;
    const int lane = tid & 31;
    const int wave = tid >> 5;
    const int half = lane >> 4;
    const int l16  = lane & 15;
    const int bh = blockIdx.y;
    const int b  = bh >> 4, h = bh & 15;
    const int qbase = blockIdx.x * 128 + wave * 16;

    const u16* Qh = Qb  + (size_t)bh * SEQ * DKH;
    const u16* Kh = Kb  + (size_t)bh * SEQ * DKH;
    const u16* Vh = Vtb + (size_t)bh * DKH * SEQ;
    const int* mrow = mask + b * SEQ;

    // staging coordinates: one 16B chunk per thread per tile
    const int krow_s = tid >> 3, kcol_s = (tid & 7) * 8;   // K tile: 32 rows x 128B
    const int vrow_s = tid >> 2, vcol_s = (tid & 3) * 8;   // V tile: 64 rows x 64B

    // resident Q fragments (K-dim 0..31, 32..63)
    FragBF a0, a1;
    const u16* qrow = Qh + (size_t)(qbase + l16) * DKH;
    a0.q[0] = *(const u32x4*)(qrow + half * 16);
    a0.q[1] = *(const u32x4*)(qrow + half * 16 + 8);
    a1.q[0] = *(const u32x4*)(qrow + 32 + half * 16);
    a1.q[1] = *(const u32x4*)(qrow + 32 + half * 16 + 8);

    float mi[8], li[8];
    v8f acc[4];
#pragma unroll
    for (int v = 0; v < 8; ++v) { mi[v] = -1e38f; li[v] = 0.0f; }
#pragma unroll
    for (int f = 0; f < 4; ++f) acc[f] = {};

    const float scale = 0.125f;   // 1/sqrt(64)

    for (int kt = 0; kt < SEQ; kt += 32) {
        // ---- cooperative K/V staging (coalesced global_load_b128 -> ds_store_b128) ----
        *(u32x4*)(&Ks[krow_s][kcol_s]) =
            *(const u32x4*)(Kh + (size_t)(kt + krow_s) * DKH + kcol_s);
        *(u32x4*)(&Vs[vrow_s][vcol_s]) =
            *(const u32x4*)(Vh + (size_t)vrow_s * SEQ + kt + vcol_s);
        __syncthreads();

        // ---- S = Q K^T (32 keys) from LDS ----
        v8f s[2];
#pragma unroll
        for (int sub = 0; sub < 2; ++sub) {
            FragBF kb0, kb1;
            const u16* kr = &Ks[sub * 16 + l16][0];
            kb0.q[0] = *(const u32x4*)(kr + half * 16);
            kb0.q[1] = *(const u32x4*)(kr + half * 16 + 8);
            kb1.q[0] = *(const u32x4*)(kr + 32 + half * 16);
            kb1.q[1] = *(const u32x4*)(kr + 32 + half * 16 + 8);
            v8f c = {};
            c = wmma_bf16(a0, kb0, c);
            c = wmma_bf16(a1, kb1, c);
            int mv = mrow[kt + sub * 16 + l16];
#pragma unroll
            for (int v = 0; v < 8; ++v) s[sub][v] = mv ? NEGV : c[v] * scale;
        }

        // ---- online softmax (row = half-wave; reduce over 16 lanes) ----
        float al[8];
#pragma unroll
        for (int v = 0; v < 8; ++v) {
            float t = fmaxf(s[0][v], s[1][v]);
            t = fmaxf(t, __shfl_xor(t, 1, 32));
            t = fmaxf(t, __shfl_xor(t, 2, 32));
            t = fmaxf(t, __shfl_xor(t, 4, 32));
            t = fmaxf(t, __shfl_xor(t, 8, 32));
            float mx = fmaxf(mi[v], t);
            al[v] = __expf(mi[v] - mx);
            float p0 = __expf(s[0][v] - mx);
            float p1 = __expf(s[1][v] - mx);
            s[0][v] = p0; s[1][v] = p1;
            float r = p0 + p1;
            r += __shfl_xor(r, 1, 32);
            r += __shfl_xor(r, 2, 32);
            r += __shfl_xor(r, 4, 32);
            r += __shfl_xor(r, 8, 32);
            li[v] = li[v] * al[v] + r;
            mi[v] = mx;
        }
#pragma unroll
        for (int f = 0; f < 4; ++f)
#pragma unroll
            for (int v = 0; v < 8; ++v) acc[f][v] *= al[v];

        // ---- P: C layout -> bf16 A layout via LDS ----
#pragma unroll
        for (int v = 0; v < 8; ++v) {
            pst[wave][half * 8 + v][l16]      = f2bf(s[0][v]);
            pst[wave][half * 8 + v][l16 + 16] = f2bf(s[1][v]);
        }
        __syncthreads();
        FragBF ap;
        ap.q[0] = *(const u32x4*)(&pst[wave][l16][half * 16]);
        ap.q[1] = *(const u32x4*)(&pst[wave][l16][half * 16 + 8]);

        // ---- O += P V from LDS ----
#pragma unroll
        for (int f = 0; f < 4; ++f) {
            FragBF bv;
            const u16* vr = &Vs[f * 16 + l16][half * 16];
            bv.q[0] = *(const u32x4*)(vr);
            bv.q[1] = *(const u32x4*)(vr + 8);
            acc[f] = wmma_bf16(ap, bv, acc[f]);
        }
        __syncthreads();   // all LDS reads done before next tile overwrites
    }

    // ---- epilogue: normalize, scatter to attn-out bf16 [b*SEQ + l][1024] ----
#pragma unroll
    for (int v = 0; v < 8; ++v) {
        float inv = 1.0f / li[v];
        int l = qbase + half * 8 + v;
        u16* orow = AO + ((size_t)(b * SEQ + l)) * DM + h * DKH;
#pragma unroll
        for (int f = 0; f < 4; ++f)
            orow[f * 16 + l16] = f2bf(acc[f][v] * inv);
    }
}

// ---------------- launcher ----------------
extern "C" void kernel_launch(void* const* d_in, const int* in_sizes, int n_in,
                              void* d_out, int out_size, void* d_ws, size_t ws_size,
                              hipStream_t stream) {
    const float* x    = (const float*)d_in[0];
    const int*   mask = (const int*)  d_in[1];
    const float* Wq   = (const float*)d_in[2];
    const float* bq   = (const float*)d_in[3];
    const float* Wk   = (const float*)d_in[4];
    const float* bk   = (const float*)d_in[5];
    const float* Wv   = (const float*)d_in[6];
    const float* bv   = (const float*)d_in[7];
    const float* Wfc  = (const float*)d_in[8];
    const float* bfc  = (const float*)d_in[9];
    float* out = (float*)d_out;

    char* ws = (char*)d_ws;
    size_t off = 0;
    const size_t ACT_B = (size_t)ROWS * DM * sizeof(u16);   // 16 MB
    const size_t W_B   = (size_t)DM * DM * sizeof(u16);     //  2 MB
    u16* xb   = (u16*)(ws + off); off += ACT_B;
    u16* wqt  = (u16*)(ws + off); off += W_B;
    u16* wkt  = (u16*)(ws + off); off += W_B;
    u16* wvt  = (u16*)(ws + off); off += W_B;
    u16* wfct = (u16*)(ws + off); off += W_B;
    u16* Qb   = (u16*)(ws + off); off += ACT_B;
    u16* Kb   = (u16*)(ws + off); off += ACT_B;
    u16* Vtb  = (u16*)(ws + off); off += ACT_B;
    u16* AOb  = (u16*)(ws + off); off += ACT_B;   // total 88 MB

    // 1) convert activations + weights to bf16 (weights transposed to [N][K])
    int nact = ROWS * DM;
    cvt_bf16_kernel<<<nact / 256, 256, 0, stream>>>(x, xb, nact);
    transpose_w_kernel<<<(DM * DM) / 256, 256, 0, stream>>>(Wq,  wqt);
    transpose_w_kernel<<<(DM * DM) / 256, 256, 0, stream>>>(Wk,  wkt);
    transpose_w_kernel<<<(DM * DM) / 256, 256, 0, stream>>>(Wv,  wvt);
    transpose_w_kernel<<<(DM * DM) / 256, 256, 0, stream>>>(Wfc, wfct);

    // 2) QKV projections (WMMA bf16, 32x64 wave tiles, double-buffered)
    dim3 ggrid(ROWS / 128, DM / 128);
    gemm_kernel<<<ggrid, 256, 0, stream>>>(xb, wqt, bq, Qb,  0);
    gemm_kernel<<<ggrid, 256, 0, stream>>>(xb, wkt, bk, Kb,  1);
    gemm_kernel<<<ggrid, 256, 0, stream>>>(xb, wvt, bv, Vtb, 2);

    // 3) flash attention over 64 (b,h) heads, K/V staged in LDS
    flash_kernel<<<dim3(SEQ / 128, BS * NH), 256, 0, stream>>>(Qb, Kb, Vtb, mask, AOb);

    // 4) final FC -> fp32 output
    gemm_kernel<<<ggrid, 256, 0, stream>>>(AOb, wfct, bfc, out, 3);
}
